// CorrelationMatchingModule_90701119357448
// MI455X (gfx1250) — matrix operand order repaired and checked by reading, not compiled, not run on hardware
//
#include <hip/hip_runtime.h>
#include <hip/hip_bf16.h>

// Problem constants (match reference)
#define BB   2
#define CC   256
#define HH   64
#define WW   64
#define HW   (HH * WW)          // 4096
#define KWIN 49
#define PAD  3
#define HP   (HH + 6)           // 70 padded rows  (y+dy in [-3, 66])
#define WP   (WW + 16)          // 80 padded cols  (WMMA #2 B-frag reads up to x0+28 -> xx=79)

typedef _Float16 half8 __attribute__((ext_vector_type(8)));
typedef _Float16 v16h  __attribute__((ext_vector_type(16)));
typedef float    v8f   __attribute__((ext_vector_type(8)));

// ---------------------------------------------------------------------------
// Kernel 1: coalesced LDS tile transpose + f32->f16 convert.
//   src [B,C,HW] f32 (channel-major)  ->  dst [B,HW,C] f16 (pixel-major)
// 32x32 tiles, 256 threads (32x8), LDS pad 33 to kill bank conflicts.
// Reads coalesced over n, writes coalesced over c.
// ---------------------------------------------------------------------------
__global__ void __launch_bounds__(256)
k_transpose_f16(const float* __restrict__ src, _Float16* __restrict__ dst) {
    __shared__ float tile[32][33];
    const int blk = blockIdx.x;            // b * (8*128) + ct * 128 + nt
    const int b  = blk >> 10;              // / 1024
    const int ct = (blk >> 7) & 7;         // c tile (8 tiles of 32)
    const int nt = blk & 127;              // n tile (128 tiles of 32)
    const int c0 = ct << 5;
    const int n0 = nt << 5;
    const int tx = threadIdx.x & 31;
    const int ty = threadIdx.x >> 5;       // 0..7

#pragma unroll
    for (int i = 0; i < 4; ++i) {
        const int c = c0 + ty + (i << 3);
        tile[ty + (i << 3)][tx] = src[((size_t)(b * CC + c) << 12) + n0 + tx];
    }
    __syncthreads();
#pragma unroll
    for (int i = 0; i < 4; ++i) {
        const int n = n0 + ty + (i << 3);
        dst[((size_t)(b * HW + n)) * CC + c0 + tx] =
            (_Float16)tile[tx][ty + (i << 3)];
    }
}

// ---------------------------------------------------------------------------
// Kernel 2: warp f2 by c1 (RAFT bilinear, zero padding) from pixel-major f2t:
// all 4 corner gathers are contiguous 512B reads per pixel-block, output is a
// contiguous 512B write. Result: f2wp [B][HP][WP][C] f16, zero-padded ring so
// the correlation kernel is branch-free (matches reference zero-pad exactly).
// ---------------------------------------------------------------------------
__device__ __forceinline__ float corner_val(const _Float16* __restrict__ f2t,
                                            int b, int xi, int yi, int c,
                                            float wt) {
    const bool valid = (xi >= 0) && (xi < WW) && (yi >= 0) && (yi < HH);
    const int xc = xi < 0 ? 0 : (xi > WW - 1 ? WW - 1 : xi);
    const int yc = yi < 0 ? 0 : (yi > HH - 1 ? HH - 1 : yi);
    const float v = (float)f2t[((size_t)(b * HW + yc * WW + xc)) * CC + c];
    return valid ? v * wt : 0.0f;
}

__global__ void __launch_bounds__(CC)
k_warp_f2(const _Float16* __restrict__ f2t, const float* __restrict__ c1,
          _Float16* __restrict__ f2wp) {
    const int p   = blockIdx.x;            // b*HP*WP + yy*WP + xx
    const int c   = threadIdx.x;
    const int b   = p / (HP * WP);
    const int rem = p - b * HP * WP;
    const int yy  = rem / WP;
    const int xx  = rem - yy * WP;
    const int y   = yy - PAD;
    const int x   = xx - PAD;
    const size_t oidx = (size_t)p * CC + c;

    if (x < 0 || x >= WW || y < 0 || y >= HH) {    // padding ring -> zeros
        f2wp[oidx] = (_Float16)0.0f;
        return;
    }
    const int n = y * WW + x;
    const float cx = c1[((size_t)b * 2 + 0) * HW + n];
    const float cy = c1[((size_t)b * 2 + 1) * HW + n];
    const float x0f = floorf(cx), y0f = floorf(cy);
    const float fx = cx - x0f, fy = cy - y0f;
    const int ix0 = (int)x0f, iy0 = (int)y0f;
    const int ix1 = ix0 + 1,  iy1 = iy0 + 1;
    const float w00 = (1.0f - fx) * (1.0f - fy);
    const float w01 = fx * (1.0f - fy);
    const float w10 = (1.0f - fx) * fy;
    const float w11 = fx * fy;

    const float acc = corner_val(f2t, b, ix0, iy0, c, w00)
                    + corner_val(f2t, b, ix1, iy0, c, w01)
                    + corner_val(f2t, b, ix0, iy1, c, w10)
                    + corner_val(f2t, b, ix1, iy1, c, w11);
    f2wp[oidx] = (_Float16)acc;
}

// ---------------------------------------------------------------------------
// Kernel 3: windowed correlation via v_wmma_f32_16x16x32_f16.
// One wave32 per (b, row y, 16-pixel tile x0). Per dy: preload ALL 16
// B-fragments (one 32-load clause), then a back-to-back 16-WMMA chain —
// no per-WMMA s_wait_loadcnt lockstep. Prefetch next padded row meanwhile.
// Two 16x16 accumulators cover the 22-wide window; K=256 via 8 chained K=32.
// ---------------------------------------------------------------------------
__global__ void __launch_bounds__(256)
k_corr_wmma(const _Float16* __restrict__ f1t,
            const _Float16* __restrict__ f2wp,
            float* __restrict__ corr) {
    const int lane = threadIdx.x & 31;
    const int wave = blockIdx.x * (blockDim.x >> 5) + (threadIdx.x >> 5);
    // 512 waves: b in {0,1}, 64 rows, 4 tiles/row
    const int b  = wave >> 8;
    const int t  = wave & 255;
    const int y  = t >> 2;
    const int x0 = (t & 3) << 4;

    const int  mA = lane & 15;           // A-frag: this lane's matrix row (pixel)
    const bool hiLane = lane >= 16;

    // --- Preload A fragments (f1 tile, reused across all 7 dy) -------------
    // A-frag order chosen here: lanes 0-15 hold K = 0..15, lanes 16-31
    // hold K = 16..31 (contiguous, matching B). Two 16B loads per frag.
    const _Float16* aptr =
        f1t + ((size_t)b * HW + (size_t)(y * WW + x0 + mA)) * CC;
    v16h A[8];
#pragma unroll
    for (int kc = 0; kc < 8; ++kc) {
        const half8 lo = *(const half8*)(aptr + kc * 32 + (hiLane ? 16 : 0));
        const half8 hi = *(const half8*)(aptr + kc * 32 + 8 + (hiLane ? 16 : 0));
        A[kc] = __builtin_shufflevector(lo, hi, 0, 1, 2, 3, 4, 5, 6, 7,
                                                8, 9, 10, 11, 12, 13, 14, 15);
    }

    // B-frag: column n = lane&15; lanes 0-15 hold K=0..15, lanes 16-31 hold
    // K=16..31 -> one contiguous 32B load per fragment (pixel-major f2wp).
    const int nB = lane & 15;
    const int kb = hiLane ? 16 : 0;

    const float scale = 0.0625f;         // 1/sqrt(256)

    for (int d = 0; d < 7; ++d) {        // dy = d - 3, padded row = y + d
        const _Float16* rrow = f2wp + ((size_t)(b * HP + (y + d)) * WP) * CC;
        const _Float16* bp1  = rrow + (size_t)(x0 + nB) * CC + kb;       // cols x0-3..x0+12
        const _Float16* bp2  = rrow + (size_t)(x0 + 16 + nB) * CC + kb;  // cols x0+13..x0+28

        // Prefetch next padded row's fragments while this row computes.
        if (d < 6) {
            const _Float16* nrow = rrow + (size_t)WP * CC;
            __builtin_prefetch(nrow + (size_t)(x0 + nB) * CC + kb, 0, 3);
            __builtin_prefetch(nrow + (size_t)(x0 + nB) * CC + kb + 128, 0, 3);
            __builtin_prefetch(nrow + (size_t)(x0 + 16 + nB) * CC + kb, 0, 3);
            __builtin_prefetch(nrow + (size_t)(x0 + 16 + nB) * CC + kb + 128, 0, 3);
        }

        // Stage all B fragments for this dy (one load clause), then WMMA chain.
        v16h B1[8], B2[8];
#pragma unroll
        for (int kc = 0; kc < 8; ++kc) {
            B1[kc] = *(const v16h*)(bp1 + kc * 32);
            B2[kc] = *(const v16h*)(bp2 + kc * 32);
        }

        v8f acc1 = {0.f, 0.f, 0.f, 0.f, 0.f, 0.f, 0.f, 0.f};
        v8f acc2 = {0.f, 0.f, 0.f, 0.f, 0.f, 0.f, 0.f, 0.f};
#pragma unroll
        for (int kc = 0; kc < 8; ++kc) {
            acc1 = __builtin_amdgcn_wmma_f32_16x16x32_f16(
                false, A[kc], false, B1[kc], (short)0, acc1, false, false);
            acc2 = __builtin_amdgcn_wmma_f32_16x16x32_f16(
                false, A[kc], false, B2[kc], (short)0, acc2, false, false);
        }

        // Scatter the 7 valid dx per pixel out of the two 16x16 products.
        // D layout: VGPR r, lanes 0-15 -> M=r, N=lane; lanes 16-31 -> M=r+8.
        // acc1[m][n]: dx+3 = n - m; acc2[m][n]: dx+3 = n + 16 - m.
#pragma unroll
        for (int r = 0; r < 8; ++r) {
            const int mm = hiLane ? (r + 8) : r;
            const int px = y * WW + x0 + mm;
            const int j1 = nB - mm;
            if (j1 >= 0 && j1 <= 6) {
                const int k = d * 7 + j1;
                corr[(size_t)(b * KWIN + k) * HW + px] = acc1[r] * scale;
            }
            const int j2 = nB + 16 - mm;
            if (j2 >= 0 && j2 <= 6) {
                const int k = d * 7 + j2;
                corr[(size_t)(b * KWIN + k) * HW + px] = acc2[r] * scale;
            }
        }
    }
}

// ---------------------------------------------------------------------------
extern "C" void kernel_launch(void* const* d_in, const int* in_sizes, int n_in,
                              void* d_out, int out_size, void* d_ws, size_t ws_size,
                              hipStream_t stream) {
    (void)in_sizes; (void)n_in; (void)out_size; (void)ws_size;

    const float* f1 = (const float*)d_in[0];   // [B,C,H,W]
    const float* f2 = (const float*)d_in[1];   // [B,C,H,W]
    const float* c1 = (const float*)d_in[2];   // [B,2,H,W]
    float* out = (float*)d_out;                // [c1 | corr]

    // Workspace partition:
    //   f1t : [B*HW][C] f16 = 4 MiB     (pixel-major f1)
    //   f2t : [B*HW][C] f16 = 4 MiB     (pixel-major f2, for coalesced gathers)
    //   f2wp: [B][HP][WP][C] f16 ~ 5.47 MiB (warped + zero-padded)
    _Float16* f1t  = (_Float16*)d_ws;
    _Float16* f2t  = (_Float16*)((char*)d_ws + ((size_t)4 << 20));
    _Float16* f2wp = (_Float16*)((char*)d_ws + ((size_t)8 << 20));

    // Output 0: c1 passthrough (stream-ordered d2d copy, graph-capture safe)
    hipMemcpyAsync(out, c1, (size_t)BB * 2 * HW * sizeof(float),
                   hipMemcpyDeviceToDevice, stream);

    // Output 1: corr at offset B*2*HW
    float* corr = out + (size_t)BB * 2 * HW;

    const int tr_blocks = BB * 8 * 128;        // (B) x (C/32) x (HW/32)
    k_transpose_f16<<<tr_blocks, 256, 0, stream>>>(f1, f1t);
    k_transpose_f16<<<tr_blocks, 256, 0, stream>>>(f2, f2t);
    k_warp_f2<<<BB * HP * WP, CC, 0, stream>>>(f2t, c1, f2wp);
    k_corr_wmma<<<64, 256, 0, stream>>>(f1t, f2wp, corr);
}
